// FFNGlobalContextExpertFusion_49469433315518
// MI455X (gfx1250) — matrix operand order repaired and checked by reading, not compiled, not run on hardware
//
#include <hip/hip_runtime.h>
#include <hip/hip_bf16.h>
#include <math.h>

// Problem constants (match reference)
#define BS   4096     // B*S tokens (M dimension)
#define DD   1024     // model dim
#define HH   4096     // hidden dim
#define EE   8        // experts
#define TILE 128      // workgroup tile (M and N)
#define KC   32       // K chunk per WMMA step (bf16 16x16x32)
#define LDSS 40       // LDS row stride in bf16 elems (80B: 16B aligned, conflict-free b128 reads)

#define AS1 __attribute__((address_space(1)))
#define AS3 __attribute__((address_space(3)))

#if defined(__gfx1250__) && __has_builtin(__builtin_amdgcn_global_load_async_to_lds_b128) && __has_builtin(__builtin_amdgcn_s_wait_asynccnt)
#define HAVE_ASYNC_LDS 1
#else
#define HAVE_ASYNC_LDS 0
#endif

typedef __attribute__((ext_vector_type(16))) __bf16 bf16x16;
typedef __attribute__((ext_vector_type(8)))  __bf16 bf16x8;
typedef __attribute__((ext_vector_type(8)))  float  v8f;
typedef int v4i __attribute__((vector_size(16)));   // matches async-LDS builtin param type

// ---------------- fragment helpers (ISA 7.12.2 layouts, wave32) ----------------

// A fragment: 16x32 bf16. lanes 0-15: M=lane, K in [0..7],[16..23];
// lanes 16-31: M=lane-16, K in [8..15],[24..31]. Two 16B LDS reads.
__device__ __forceinline__ bf16x16 frag_a(const __bf16* lds, int mBase, int lane) {
    const __bf16* p = lds + (mBase + (lane & 15)) * LDSS + ((lane >> 4) << 3);
    bf16x8 lo = *(const bf16x8*)(p);
    bf16x8 hi = *(const bf16x8*)(p + 16);
    return __builtin_shufflevector(lo, hi, 0,1,2,3,4,5,6,7,8,9,10,11,12,13,14,15);
}

// B fragment: 32x16 bf16 staged transposed in LDS as [n][k].
// lane L: N = L%16, K = (L/16)*8 + [0..7], 16 + (L/16)*8 + [0..7] -> two b128s.
__device__ __forceinline__ bf16x16 frag_b(const __bf16* lds, int nBase, int lane) {
    const __bf16* p = lds + (nBase + (lane & 15)) * LDSS + ((lane >> 4) << 3);
    bf16x8 lo = *(const bf16x8*)(p);
    bf16x8 hi = *(const bf16x8*)(p + 16);
    return __builtin_shufflevector(lo, hi, 0,1,2,3,4,5,6,7,8,9,10,11,12,13,14,15);
}

__device__ __forceinline__ void do_wmma(const __bf16* Asb, const __bf16* Bsb,
                                        int mWave, int nWave, int lane,
                                        v8f (&acc)[2][4]) {
    bf16x16 af[2], bfv[4];
#pragma unroll
    for (int mi = 0; mi < 2; ++mi) af[mi] = frag_a(Asb, mWave + mi * 16, lane);
#pragma unroll
    for (int ni = 0; ni < 4; ++ni) bfv[ni] = frag_b(Bsb, nWave + ni * 16, lane);
#pragma unroll
    for (int mi = 0; mi < 2; ++mi)
#pragma unroll
        for (int ni = 0; ni < 4; ++ni)
            acc[mi][ni] = __builtin_amdgcn_wmma_f32_16x16x32_bf16(
                false, af[mi], false, bfv[ni], (short)0, acc[mi][ni], false, false);
}

__device__ __forceinline__ float gelu_exact(float v) {
    return 0.5f * v * (1.0f + erff(v * 0.70710678118654752f));
}

// ---------------- staging helpers (register-staged, for ping-pong overlap) ----------------

// A tile (f32 source): 128x32, thread covers rows t>>3 (+p*32), cols (t&7)*4..+3
__device__ __forceinline__ void a32_loads(const float* a, int ld, int tileM, int kb,
                                          int t, float4 (&ra)[4]) {
    const int kk = (t & 7) * 4, row = t >> 3;
#pragma unroll
    for (int p = 0; p < 4; ++p)
        ra[p] = *(const float4*)(a + (size_t)(tileM + row + p * 32) * ld + kb + kk);
}
__device__ __forceinline__ void a32_store(__bf16* Asb, int t, const float4 (&ra)[4]) {
    const int kk = (t & 7) * 4, row = t >> 3;
#pragma unroll
    for (int p = 0; p < 4; ++p) {
        __bf16* dst = Asb + (row + p * 32) * LDSS + kk;
        dst[0] = (__bf16)ra[p].x; dst[1] = (__bf16)ra[p].y;
        dst[2] = (__bf16)ra[p].z; dst[3] = (__bf16)ra[p].w;
    }
}

// W tile (f32 source, k-major): 32x128, thread covers k = t>>5 (+p*8), n = (t&31)*4..+3
__device__ __forceinline__ void w_loads(const float* w, int ld, int tileN, int kb,
                                        int t, float4 (&rb)[4]) {
    const int nb = (t & 31) * 4, kr = t >> 5;
#pragma unroll
    for (int p = 0; p < 4; ++p) {
        const float* src = w + (size_t)(kb + kr + p * 8) * ld + tileN + nb;
        if (p == 0) __builtin_prefetch(src + (size_t)KC * ld, 0, 0); // global_prefetch
        rb[p] = *(const float4*)src;
    }
}
__device__ __forceinline__ void w_store(__bf16* Bsb, int t, const float4 (&rb)[4]) {
    const int nb = (t & 31) * 4, kr = t >> 5;
#pragma unroll
    for (int p = 0; p < 4; ++p) {
        const int k = kr + p * 8;
        Bsb[(nb + 0) * LDSS + k] = (__bf16)rb[p].x;
        Bsb[(nb + 1) * LDSS + k] = (__bf16)rb[p].y;
        Bsb[(nb + 2) * LDSS + k] = (__bf16)rb[p].z;
        Bsb[(nb + 3) * LDSS + k] = (__bf16)rb[p].w;
    }
}

// A tile (bf16 source = h): straight 32B copy per thread; async global->LDS if available.
__device__ __forceinline__ void abf_stage(const __bf16* hA, __bf16* Asb, int kb, int t) {
    const int row = t >> 1, seg = (t & 1) * 16;
    const __bf16* src = hA + (size_t)row * HH + kb + seg;
    __bf16* dst = Asb + row * LDSS + seg;
#if HAVE_ASYNC_LDS
    __builtin_amdgcn_global_load_async_to_lds_b128(
        (AS1 v4i*)src, (AS3 v4i*)dst, 0, 0);
    __builtin_amdgcn_global_load_async_to_lds_b128(
        (AS1 v4i*)(src + 8), (AS3 v4i*)(dst + 8), 0, 0);
#else
    ((float4*)dst)[0] = ((const float4*)src)[0];
    ((float4*)dst)[1] = ((const float4*)src)[1];
#endif
}
__device__ __forceinline__ void abf_wait() {
#if HAVE_ASYNC_LDS
    __builtin_amdgcn_s_wait_asynccnt(0);
#endif
}

// ---------------- kernel 0: top-2 router ----------------
__global__ void moe_router(const float* __restrict__ scores,
                           float* __restrict__ masked,
                           float* __restrict__ counts) {
    int tok = blockIdx.x * blockDim.x + threadIdx.x;
    if (tok >= BS) return;
    const float* s = scores + tok * EE;
    float v[EE];
#pragma unroll
    for (int e = 0; e < EE; ++e) v[e] = s[e];
    int i1 = 0; float m1 = v[0];
#pragma unroll
    for (int e = 1; e < EE; ++e) if (v[e] > m1) { m1 = v[e]; i1 = e; }
    int i2 = -1; float m2 = -3.4e38f;
#pragma unroll
    for (int e = 0; e < EE; ++e) if (e != i1 && v[e] > m2) { m2 = v[e]; i2 = e; }
    float scale = 1.0f / (m1 + m2 + 1e-8f);
#pragma unroll
    for (int e = 0; e < EE; ++e)
        masked[tok * EE + e] = (e == i1 || e == i2) ? scale * v[e] : 0.0f;
    atomicAdd(&counts[i1], 1.0f);
    atomicAdd(&counts[i2], 1.0f);
}

// ---------------- kernel 1: h = gelu(x @ W1[e] + b1[e]) -> bf16 ----------------
// grid (HH/TILE, BS/TILE, EE), block 256 (8 waves; wave tile 32x64 = 2x4 wmma frags)
// launch_bounds(256,1): 8 waves/WGP is plenty; give the allocator a full VGPR
// budget so the 64 accumulator VGPRs + staged tiles never spill to scratch.
__global__ void __launch_bounds__(256, 1)
moe_gemm1(const float* __restrict__ x,
          const float* __restrict__ W1,
          const float* __restrict__ b1,
          __bf16* __restrict__ h) {
    __shared__ __bf16 As[2][TILE * LDSS];
    __shared__ __bf16 Bs[2][TILE * LDSS];
    __shared__ float  b1s[TILE];

    const int t     = threadIdx.x;
    const int lane  = t & 31;
    const int w     = t >> 5;
    const int mWave = (w & 3) * 32;
    const int nWave = (w >> 2) * 64;
    const int tileM = blockIdx.y * TILE;
    const int tileN = blockIdx.x * TILE;
    const int e     = blockIdx.z;
    const float* w1e = W1 + (size_t)e * DD * HH;

    if (t < TILE) b1s[t] = b1[e * HH + tileN + t];

    v8f acc[2][4];
#pragma unroll
    for (int mi = 0; mi < 2; ++mi)
#pragma unroll
        for (int ni = 0; ni < 4; ++ni)
            acc[mi][ni] = (v8f){0,0,0,0,0,0,0,0};

    // prologue: stage kb=0 into buffer 0
    {
        float4 ra[4], rb[4];
        a32_loads(x, DD, tileM, 0, t, ra);
        w_loads(w1e, HH, tileN, 0, t, rb);
        a32_store(As[0], t, ra);
        w_store(Bs[0], t, rb);
    }

    int buf = 0;
    for (int kb = 0; kb < DD; kb += KC) {
        __syncthreads();                      // publish staged buffer
        const bool next = (kb + KC) < DD;
        float4 ra[4], rb[4];
        if (next) {                           // issue next tile's loads before WMMAs
            a32_loads(x, DD, tileM, kb + KC, t, ra);
            w_loads(w1e, HH, tileN, kb + KC, t, rb);
        }
        do_wmma(As[buf], Bs[buf], mWave, nWave, lane, acc);
        if (next) {                           // convert + store after WMMAs issued
            a32_store(As[buf ^ 1], t, ra);
            w_store(Bs[buf ^ 1], t, rb);
        }
        buf ^= 1;
    }

    // epilogue: bias + exact GELU + bf16 store to h[e][m][n]
    const int lm = (lane >> 4) * 8;
    const int ln = lane & 15;
#pragma unroll
    for (int mi = 0; mi < 2; ++mi)
#pragma unroll
        for (int ni = 0; ni < 4; ++ni) {
            const int nLoc = nWave + ni * 16 + ln;
            const float bb = b1s[nLoc];
            __bf16* dst = h + ((size_t)e * BS + tileM + mWave + mi * 16 + lm) * HH
                            + tileN + nLoc;
#pragma unroll
            for (int r = 0; r < 8; ++r)
                dst[(size_t)r * HH] = (__bf16)gelu_exact(acc[mi][ni][r] + bb);
        }
}

// ---------------- kernel 2: out = sum_e s[m,e] * (h_e @ W2[e] + b2[e]) ----------------
// grid (DD/TILE, BS/TILE), block 256; expert loop inside.
// Two accumulator sets (128 VGPRs) -> definitely needs the full VGPR budget.
__global__ void __launch_bounds__(256, 1)
moe_gemm2(const __bf16* __restrict__ h,
          const float* __restrict__ W2,
          const float* __restrict__ b2,
          const float* __restrict__ masked,
          float* __restrict__ out) {
    __shared__ __bf16 As[2][TILE * LDSS];
    __shared__ __bf16 Bs[2][TILE * LDSS];
    __shared__ float  sS[TILE * EE];
    __shared__ float  b2s[EE * TILE];

    const int t     = threadIdx.x;
    const int lane  = t & 31;
    const int w     = t >> 5;
    const int mWave = (w & 3) * 32;
    const int nWave = (w >> 2) * 64;
    const int tileM = blockIdx.y * TILE;
    const int tileN = blockIdx.x * TILE;

#pragma unroll
    for (int j = 0; j < 4; ++j) sS[t * 4 + j] = masked[(size_t)tileM * EE + t * 4 + j];
#pragma unroll
    for (int j = 0; j < 4; ++j) {
        const int idx = t + j * 256;
        b2s[idx] = b2[(idx >> 7) * DD + tileN + (idx & 127)];
    }

    v8f accF[2][4];
#pragma unroll
    for (int mi = 0; mi < 2; ++mi)
#pragma unroll
        for (int ni = 0; ni < 4; ++ni)
            accF[mi][ni] = (v8f){0,0,0,0,0,0,0,0};

    const int lm = (lane >> 4) * 8;
    const int ln = lane & 15;

    for (int e = 0; e < EE; ++e) {
        const __bf16* hA  = h  + ((size_t)e * BS + tileM) * HH;
        const float*  w2e = W2 + (size_t)e * HH * DD;
        v8f acc[2][4];
#pragma unroll
        for (int mi = 0; mi < 2; ++mi)
#pragma unroll
            for (int ni = 0; ni < 4; ++ni)
                acc[mi][ni] = (v8f){0,0,0,0,0,0,0,0};

        // prologue for this expert: stage kb=0 into buffer 0
        {
            float4 rb[4];
            abf_stage(hA, As[0], 0, t);                 // async (or fallback) bf16 copy
            w_loads(w2e, DD, tileN, 0, t, rb);
            w_store(Bs[0], t, rb);
            abf_wait();
        }

        int buf = 0;
        for (int kb = 0; kb < HH; kb += KC) {
            __syncthreads();
            const bool next = (kb + KC) < HH;
            float4 rb[4];
            if (next) {
                abf_stage(hA, As[buf ^ 1], kb + KC, t); // async issue, no VGPR roundtrip
                w_loads(w2e, DD, tileN, kb + KC, t, rb);
            }
            do_wmma(As[buf], Bs[buf], mWave, nWave, lane, acc);
            if (next) {
                w_store(Bs[buf ^ 1], t, rb);
                abf_wait();                             // async done before next publish
            }
            buf ^= 1;
        }

        // scale per-expert result by routing weight, accumulate
#pragma unroll
        for (int mi = 0; mi < 2; ++mi) {
            const int m0 = mWave + mi * 16 + lm;
#pragma unroll
            for (int ni = 0; ni < 4; ++ni) {
                const int nLoc = nWave + ni * 16 + ln;
                const float bb = b2s[e * TILE + nLoc];
#pragma unroll
                for (int r = 0; r < 8; ++r) {
                    const float sc = sS[(m0 + r) * EE + e];
                    accF[mi][ni][r] += sc * (acc[mi][ni][r] + bb);
                }
            }
        }
    }

    // final store
#pragma unroll
    for (int mi = 0; mi < 2; ++mi) {
        const int m0 = tileM + mWave + mi * 16 + lm;
#pragma unroll
        for (int ni = 0; ni < 4; ++ni) {
            const int nG = tileN + nWave + ni * 16 + ln;
#pragma unroll
            for (int r = 0; r < 8; ++r)
                out[(size_t)(m0 + r) * DD + nG] = accF[mi][ni][r];
        }
    }
}

// ---------------- launch ----------------
// Workspace layout: [0, 128KB) masked routing scores (BS*EE f32)
//                   [128KB, 128KB + 256MB) h in bf16 (EE*BS*HH)
extern "C" void kernel_launch(void* const* d_in, const int* in_sizes, int n_in,
                              void* d_out, int out_size, void* d_ws, size_t ws_size,
                              hipStream_t stream) {
    const float* x      = (const float*)d_in[0];
    const float* scores = (const float*)d_in[1];
    const float* W1     = (const float*)d_in[2];
    const float* b1     = (const float*)d_in[3];
    const float* W2     = (const float*)d_in[4];
    const float* b2     = (const float*)d_in[5];

    float* out    = (float*)d_out;
    float* counts = out + (size_t)BS * DD;     // second tuple output (E floats)

    float*  ws_scores = (float*)d_ws;
    __bf16* ws_h      = (__bf16*)((char*)d_ws + (size_t)BS * EE * sizeof(float));

    (void)hipMemsetAsync(counts, 0, EE * sizeof(float), stream);
    moe_router<<<BS / 256, 256, 0, stream>>>(scores, ws_scores, counts);

    dim3 g1(HH / TILE, BS / TILE, EE);
    moe_gemm1<<<g1, 256, 0, stream>>>(x, W1, b1, ws_h);

    dim3 g2(DD / TILE, BS / TILE);
    moe_gemm2<<<g2, 256, 0, stream>>>(ws_h, W2, b2, ws_scores, out);
}
